// Transformer_21139829031458
// MI455X (gfx1250) — compile-verified, matrix operand order checked by reading, not run on hardware
//
#include <hip/hip_runtime.h>
#include <math.h>

typedef __attribute__((ext_vector_type(16))) _Float16 v16h;
typedef __attribute__((ext_vector_type(8)))  float    v8f;

#define DEVINL __device__ __forceinline__

// Problem constants
#define CB   8
#define CS   512
#define CMEM 512
#define CL   4
#define CE   512
#define CH   8
#define CDH  64
#define COBS 128
#define CK   1024

DEVINL v8f wmma_f16(v16h a, v16h b, v8f c) {
  return __builtin_amdgcn_wmma_f32_16x16x32_f16(false, a, false, b, (short)0, c,
                                                false, false);
}

DEVINL v8f vzero() {
  v8f z = {0.f, 0.f, 0.f, 0.f, 0.f, 0.f, 0.f, 0.f};
  return z;
}

// Load one f16 WMMA operand fragment (16x32 A-layout, or its B mirror):
// two 16-byte loads per lane at half-dependent offsets (units: halves).
DEVINL v16h load_frag(const _Float16* p, int lo, int hi) {
  union { v16h v; uint4 q[2]; } u;
  u.q[0] = *reinterpret_cast<const uint4*>(p + lo);
  u.q[1] = *reinterpret_cast<const uint4*>(p + hi);
  return u.v;
}

// ---------------------------------------------------------------------------
// One-time weight conversion: f32 [Kd,N] -> f16 transposed [N,Kd]
// ---------------------------------------------------------------------------
__global__ void __launch_bounds__(256)
convT_kernel(const float* __restrict__ in, _Float16* __restrict__ out, int Kd,
             int N) {
  const int i = blockIdx.x * 256 + threadIdx.x;
  const int n = i / Kd, k = i % Kd;
  out[i] = (_Float16)in[(long)k * N + n];
}

// Plain f32 -> f16 copy (obs)
__global__ void __launch_bounds__(256)
conv_kernel(const float* __restrict__ in, _Float16* __restrict__ out) {
  const int i = blockIdx.x * 256 + threadIdx.x;
  out[i] = (_Float16)in[i];
}

// ---------------------------------------------------------------------------
// Tiled WMMA GEMM: C[M,N] = A[M,Kd](f16) * Bt[N,Kd](f16, pre-transposed),
// f32 accumulation. Block tile 64x64, 256 threads = 8 waves, wave 16x32.
// A tile (64x32 f16 = 4KB) is staged into LDS with the CDNA5 async DMA
// (global_load_async_to_lds_b128, ASYNCcnt). B fragments are read straight
// from global Wt rows (L2-resident weights).
// MODE 0: f16 out = acc + bias[n]                        (K proj, r proj)
// MODE 1: two f16 outs = acc + bias + u / + v            (Q proj -> qu,qv)
// MODE 2: f16 transposed out [bt][n][row%tRows]          (V proj -> V^T)
// MODE 3: f32 out = acc + bias + residual[m,n]           (Wo, FFN2)
// MODE 4: f16 out = gelu(acc + bias)                     (FFN1)
// MODE 5: f32 out = acc + bias                           (encoder)
// ---------------------------------------------------------------------------
template <int MODE>
__global__ void __launch_bounds__(256)
gemm_wmma(const _Float16* __restrict__ A, const _Float16* __restrict__ Bt,
          const float* __restrict__ bias, const float* __restrict__ ex0,
          const float* __restrict__ ex1, void* __restrict__ out0,
          void* __restrict__ out1, int M, int N, int Kd, int aBatchRows,
          int aBatchStride, int aRowOff, int tRows) {
  __shared__ __align__(16) _Float16 sA[64 * 32];  // [row][k], 4KB

  const int tid = threadIdx.x;
  const int lane = tid & 31, wv = tid >> 5;
  const int hf = lane >> 4, nl = lane & 15;
  const int mw = wv >> 1, nw = wv & 1;
  const int bm = blockIdx.y * 64, bn = blockIdx.x * 64;
  const int offLo = hf * 8, offHi = 16 + hf * 8;

  v8f acc0 = vzero(), acc1 = vzero();

  // async staging coords: one b128 (16B) per thread -> 4KB tile
  const int ar = tid >> 2;            // 0..63 row in A tile
  const int aseg = (tid & 3) * 16;    // byte segment within 64B row
  const int gAr = bm + ar;
  const long aRow =
      (long)(gAr / aBatchRows) * aBatchStride + aRowOff + (gAr % aBatchRows);
  const _Float16* aBase = A + aRow * (long)Kd;
  const unsigned ldsA =
      (unsigned)(size_t)(void*)(&sA[0]) + (unsigned)(ar * 64 + aseg);

  // B fragment row pointers (Bt row-major [N][Kd])
  const _Float16* bRow0 = Bt + (long)(bn + nw * 32 + nl) * Kd;
  const _Float16* bRow1 = Bt + (long)(bn + nw * 32 + 16 + nl) * Kd;

  for (int k0 = 0; k0 < Kd; k0 += 32) {
    __syncthreads();
    {
      const unsigned long long ga =
          (unsigned long long)(const void*)((const char*)(aBase + k0) + aseg);
      asm volatile("global_load_async_to_lds_b128 %0, %1, off" ::"v"(ldsA),
                   "v"(ga)
                   : "memory");
      asm volatile("s_wait_asynccnt 0" ::: "memory");
    }
    __syncthreads();
    if (k0 + 32 < Kd) {
      __builtin_prefetch((const void*)(bRow0 + k0 + 32), 0, 0);
      __builtin_prefetch((const void*)(bRow1 + k0 + 32), 0, 0);
    }
    v16h aF = load_frag(&sA[(mw * 16 + nl) * 32], offLo, offHi);
    v16h bF0 = load_frag(bRow0 + k0, offLo, offHi);
    v16h bF1 = load_frag(bRow1 + k0, offLo, offHi);
    acc0 = wmma_f16(aF, bF0, acc0);
    acc1 = wmma_f16(aF, bF1, acc1);
  }

  // Epilogue. C layout: VGPR g, lane -> (m = g + 8*(lane/16), n = lane%16)
  const int mB = bm + mw * 16 + 8 * hf;
#pragma unroll
  for (int sub = 0; sub < 2; ++sub) {
    const v8f acc = sub ? acc1 : acc0;
    const int n = bn + nw * 32 + sub * 16 + nl;
    const float bb = bias ? bias[n] : 0.0f;
#pragma unroll
    for (int g = 0; g < 8; ++g) {
      const int m = mB + g;
      const float v = acc[g] + bb;
      if (MODE == 0) {
        ((_Float16*)out0)[(long)m * N + n] = (_Float16)v;
      } else if (MODE == 1) {
        ((_Float16*)out0)[(long)m * N + n] = (_Float16)(v + ex0[n]);
        ((_Float16*)out1)[(long)m * N + n] = (_Float16)(v + ex1[n]);
      } else if (MODE == 2) {
        const int bt = m / tRows, rr = m % tRows;
        ((_Float16*)out0)[((long)bt * N + n) * tRows + rr] = (_Float16)v;
      } else if (MODE == 3) {
        ((float*)out0)[(long)m * N + n] = v + ex0[(long)m * N + n];
      } else if (MODE == 4) {
        const float t = 0.7978845608028654f * (v + 0.044715f * v * v * v);
        ((_Float16*)out0)[(long)m * N + n] =
            (_Float16)(0.5f * v * (1.0f + tanhf(t)));
      } else {
        ((float*)out0)[(long)m * N + n] = v;
      }
    }
  }
}

// ---------------------------------------------------------------------------
// Fused Transformer-XL flash attention.
// One wave per (b, h, 16-row q-tile), 32 keys per iteration.
// scores[q,k] = mask ? ((q+u)·K[k] + (q+v)·r[k+511-q]) * 0.125 : -1e30
// mask <=> j = k+511-q <= 1023. Rel-shift extracted with ds_bpermute.
// Online softmax; P transposed to A-layout through 1KB/wave LDS; o = P@V^T.
// ---------------------------------------------------------------------------
__global__ void __launch_bounds__(256)
attn_kernel(const _Float16* __restrict__ qu, const _Float16* __restrict__ qv,
            const _Float16* __restrict__ kf, const _Float16* __restrict__ vT,
            const _Float16* __restrict__ rf, _Float16* __restrict__ o) {
  __shared__ __align__(16) _Float16 pbuf[8 * 16 * 32];
  const int tid = threadIdx.x;
  const int lane = tid & 31, wv = tid >> 5;
  const int gw = blockIdx.x * 8 + wv;
  const int qt = gw & 31, hh = (gw >> 5) & 7, b = gw >> 8;
  const int Q0 = qt * 16;
  const int hf = lane >> 4, nl = lane & 15;
  const int offLo = hf * 8, offHi = 16 + hf * 8;
  _Float16* myP = &pbuf[wv * 16 * 32];

  // Q fragments (A layout rows = lane&15), persistent across the k loop
  const _Float16* quRow = qu + ((long)(b * CS + Q0 + nl) * CE + hh * CDH);
  const _Float16* qvRow = qv + ((long)(b * CS + Q0 + nl) * CE + hh * CDH);
  v16h aU[2], aV[2];
#pragma unroll
  for (int c = 0; c < 2; ++c) {
    aU[c] = load_frag(quRow, c * 32 + offLo, c * 32 + offHi);
    aV[c] = load_frag(qvRow, c * 32 + offLo, c * 32 + offHi);
  }

  v8f oacc[4];
  float mst[8], lst[8];
#pragma unroll
  for (int d = 0; d < 4; ++d) oacc[d] = vzero();
#pragma unroll
  for (int g = 0; g < 8; ++g) { mst[g] = -3.0e38f; lst[g] = 0.0f; }

  const int nk = ((527 + Q0) >> 5) + 1;
  for (int kt = 0; kt < nk; ++kt) {
    const int KT0 = kt * 32;
    const int jbase = KT0 + 496 - Q0;  // j = jbase + (n + 15 - m)

    // content term ac: two 16-col subtiles, Kdim = DH = 64
    v8f accS[2];
#pragma unroll
    for (int sub = 0; sub < 2; ++sub) {
      const _Float16* kRow =
          kf + ((long)(b * CK + KT0 + sub * 16 + nl) * CE + hh * CDH);
      v16h kb0 = load_frag(kRow, offLo, offHi);
      v16h kb1 = load_frag(kRow, 32 + offLo, 32 + offHi);
      accS[sub] = wmma_f16(aU[1], kb1, wmma_f16(aU[0], kb0, vzero()));
    }

    // position term bd_raw over j in [jbase, jbase+47]: three 16-col tiles
    v8f bd[3];
#pragma unroll
    for (int rt = 0; rt < 3; ++rt) {
      int j = jbase + rt * 16 + nl;
      j = j > 1023 ? 1023 : j;  // clamped rows only feed masked elements
      const _Float16* rRow = rf + ((long)j * CE + hh * CDH);
      v16h rb0 = load_frag(rRow, offLo, offHi);
      v16h rb1 = load_frag(rRow, 32 + offLo, 32 + offHi);
      bd[rt] = wmma_f16(aV[1], rb1, wmma_f16(aV[0], rb0, vzero()));
    }

    // combine + rel-shift extraction + online softmax
    float pA[8], pB[8];
#pragma unroll
    for (int g = 0; g < 8; ++g) {
      const int m = g + 8 * hf;
      const int t0 = nl + 15 - m;           // 0..30  (subtile 0)
      const int t1 = t0 + 16;               // 16..46 (subtile 1)
      const int idx = (hf * 16 + (t0 & 15)) * 4;
      const float e0 = __int_as_float(
          __builtin_amdgcn_ds_bpermute(idx, __float_as_int(bd[0][g])));
      const float e1 = __int_as_float(
          __builtin_amdgcn_ds_bpermute(idx, __float_as_int(bd[1][g])));
      const float e2 = __int_as_float(
          __builtin_amdgcn_ds_bpermute(idx, __float_as_int(bd[2][g])));
      float s0 = (accS[0][g] + ((t0 < 16) ? e0 : e1)) * 0.125f;
      float s1 = (accS[1][g] + ((t1 < 32) ? e1 : e2)) * 0.125f;
      if (jbase + t0 > 1023) s0 = -1.0e30f;
      if (jbase + t1 > 1023) s1 = -1.0e30f;

      float tm = fmaxf(s0, s1);
      tm = fmaxf(tm, __shfl_xor(tm, 1));
      tm = fmaxf(tm, __shfl_xor(tm, 2));
      tm = fmaxf(tm, __shfl_xor(tm, 4));
      tm = fmaxf(tm, __shfl_xor(tm, 8));
      const float mnew = fmaxf(mst[g], tm);
      const float corr = __expf(mst[g] - mnew);
      const float p0 = __expf(s0 - mnew);
      const float p1 = __expf(s1 - mnew);
      float rs = p0 + p1;
      rs += __shfl_xor(rs, 1);
      rs += __shfl_xor(rs, 2);
      rs += __shfl_xor(rs, 4);
      rs += __shfl_xor(rs, 8);
      lst[g] = lst[g] * corr + rs;
      mst[g] = mnew;
#pragma unroll
      for (int d = 0; d < 4; ++d) oacc[d][g] *= corr;
      pA[g] = p0;
      pB[g] = p1;
    }

    // P: C-layout -> A-layout via per-wave LDS transpose
#pragma unroll
    for (int g = 0; g < 8; ++g) {
      const int m = g + 8 * hf;
      myP[m * 32 + nl] = (_Float16)pA[g];
      myP[m * 32 + 16 + nl] = (_Float16)pB[g];
    }
    __asm__ volatile("s_wait_dscnt 0" ::: "memory");
    v16h aP = load_frag(&myP[nl * 32], offLo, offHi);

    // o += P @ V  (V^T layout: [b][feature][key], contiguous keys)
#pragma unroll
    for (int d = 0; d < 4; ++d) {
      const _Float16* vRow =
          vT + ((long)(b * CE + hh * CDH + d * 16 + nl) * CK + KT0);
      v16h vb = load_frag(vRow, offLo, offHi);
      oacc[d] = wmma_f16(aP, vb, oacc[d]);
    }
  }

  // finalize: divide by softmax denominator, write f16 o[b,q,h*64+d]
#pragma unroll
  for (int g = 0; g < 8; ++g) {
    const float inv = 1.0f / lst[g];
    const int m = Q0 + g + 8 * hf;
#pragma unroll
    for (int d = 0; d < 4; ++d) {
      o[(long)(b * CS + m) * CE + hh * CDH + d * 16 + nl] =
          (_Float16)(oacc[d][g] * inv);
    }
  }
}

// ---------------------------------------------------------------------------
// LayerNorm over E=512 of concat(memories[:,:,layer], x) -> vk [B,K,E] f16
// ---------------------------------------------------------------------------
__global__ void __launch_bounds__(256)
ln_concat_kernel(const float* __restrict__ x, const float* __restrict__ mem,
                 const float* __restrict__ sg, const float* __restrict__ bg,
                 _Float16* __restrict__ vk, int layer) {
  __shared__ float red[256];
  __shared__ float stat[2];
  const int row = blockIdx.x;  // b*1024 + kk
  const int b = row >> 10, kk = row & 1023;
  const float* src = (kk < CMEM)
      ? mem + (((long)(b * CMEM + kk) * CL + layer) * CE)
      : x + ((long)(b * CS + (kk - CMEM)) * CE);
  const int t = threadIdx.x;
  const float v0 = src[t * 2], v1 = src[t * 2 + 1];
  red[t] = v0 + v1;
  __syncthreads();
  for (int s = 128; s > 0; s >>= 1) {
    if (t < s) red[t] += red[t + s];
    __syncthreads();
  }
  if (t == 0) stat[0] = red[0] * (1.0f / 512.0f);
  __syncthreads();
  const float mu = stat[0];
  const float d0 = v0 - mu, d1 = v1 - mu;
  red[t] = d0 * d0 + d1 * d1;
  __syncthreads();
  for (int s = 128; s > 0; s >>= 1) {
    if (t < s) red[t] += red[t + s];
    __syncthreads();
  }
  if (t == 0) stat[1] = rsqrtf(red[0] * (1.0f / 512.0f) + 1e-6f);
  __syncthreads();
  const float inv = stat[1];
  _Float16* out = vk + (long)row * CE;
  out[t * 2] = (_Float16)(d0 * inv * sg[t * 2] + bg[t * 2]);
  out[t * 2 + 1] = (_Float16)(d1 * inv * sg[t * 2 + 1] + bg[t * 2 + 1]);
}

// Plain row LayerNorm (4096 rows of E=512), f32 in -> f16 out
__global__ void __launch_bounds__(256)
ln_kernel(const float* __restrict__ in, const float* __restrict__ sg,
          const float* __restrict__ bg, _Float16* __restrict__ out) {
  __shared__ float red[256];
  __shared__ float stat[2];
  const float* src = in + (long)blockIdx.x * CE;
  const int t = threadIdx.x;
  const float v0 = src[t * 2], v1 = src[t * 2 + 1];
  red[t] = v0 + v1;
  __syncthreads();
  for (int s = 128; s > 0; s >>= 1) {
    if (t < s) red[t] += red[t + s];
    __syncthreads();
  }
  if (t == 0) stat[0] = red[0] * (1.0f / 512.0f);
  __syncthreads();
  const float mu = stat[0];
  const float d0 = v0 - mu, d1 = v1 - mu;
  red[t] = d0 * d0 + d1 * d1;
  __syncthreads();
  for (int s = 128; s > 0; s >>= 1) {
    if (t < s) red[t] += red[t + s];
    __syncthreads();
  }
  if (t == 0) stat[1] = rsqrtf(red[0] * (1.0f / 512.0f) + 1e-6f);
  __syncthreads();
  const float inv = stat[1];
  _Float16* dst = out + (long)blockIdx.x * CE;
  dst[t * 2] = (_Float16)(d0 * inv * sg[t * 2] + bg[t * 2]);
  dst[t * 2 + 1] = (_Float16)(d1 * inv * sg[t * 2 + 1] + bg[t * 2 + 1]);
}

// Sinusoidal positional embedding (f16), pos_seq = [K, K-1, ..., 1]
__global__ void __launch_bounds__(256) pe_kernel(_Float16* __restrict__ pe) {
  const int i = blockIdx.x * 256 + threadIdx.x;  // 0 .. K*E-1
  const int j = i >> 9, e = i & 511;
  const int tt = e & 255;
  const float pos = (float)(CK - j);
  const float freq =
      __expf(-(float)tt * (1.0f / 256.0f) * 9.210340371976184f);  // ln(1e4)
  const float arg = pos * freq;
  pe[i] = (_Float16)((e < 256) ? __sinf(arg) : __cosf(arg));
}

// ---------------------------------------------------------------------------
// Host-side orchestration
// ---------------------------------------------------------------------------
extern "C" void kernel_launch(void* const* d_in, const int* in_sizes, int n_in,
                              void* d_out, int out_size, void* d_ws,
                              size_t ws_size, hipStream_t stream) {
  (void)in_sizes; (void)n_in; (void)out_size; (void)ws_size;

  const float* obs      = (const float*)d_in[0];
  const float* memories = (const float*)d_in[1];
  // d_in[2] = mask (recomputed analytically: k <= MEM + q)
  const float* W_enc = (const float*)d_in[3];
  const float* b_enc = (const float*)d_in[4];
  const float* ln1_s = (const float*)d_in[5];
  const float* ln1_b = (const float*)d_in[6];
  const float* Wq = (const float*)d_in[7];
  const float* bq = (const float*)d_in[8];
  const float* Wk = (const float*)d_in[9];
  const float* bk = (const float*)d_in[10];
  const float* Wv = (const float*)d_in[11];
  const float* bv = (const float*)d_in[12];
  const float* Wr = (const float*)d_in[13];
  const float* ub = (const float*)d_in[14];
  const float* vb = (const float*)d_in[15];
  const float* Wo = (const float*)d_in[16];
  const float* bo = (const float*)d_in[17];
  const float* ln2_s = (const float*)d_in[18];
  const float* ln2_b = (const float*)d_in[19];
  const float* W1 = (const float*)d_in[20];
  const float* b1 = (const float*)d_in[21];
  const float* W2 = (const float*)d_in[22];
  const float* b2 = (const float*)d_in[23];

  char* ws = (char*)d_ws;
  size_t off = 0;
  auto alloc = [&](size_t bytes) {
    void* p = ws + off;
    off += (bytes + 255) & ~(size_t)255;
    return p;
  };
  const size_t WPL = (size_t)CE * CE;  // weights per layer
  _Float16* WencT = (_Float16*)alloc((size_t)COBS * CE * 2);
  _Float16* WqT = (_Float16*)alloc(CL * WPL * 2);
  _Float16* WkT = (_Float16*)alloc(CL * WPL * 2);
  _Float16* WvT = (_Float16*)alloc(CL * WPL * 2);
  _Float16* WrT = (_Float16*)alloc(CL * WPL * 2);
  _Float16* WoT = (_Float16*)alloc(CL * WPL * 2);
  _Float16* W1T = (_Float16*)alloc(CL * WPL * 2);
  _Float16* W2T = (_Float16*)alloc(CL * WPL * 2);
  _Float16* obs_h = (_Float16*)alloc((size_t)CB * CS * COBS * 2);
  _Float16* pe = (_Float16*)alloc((size_t)CK * CE * 2);
  float*    x  = (float*)alloc((size_t)CB * CS * CE * 4);
  _Float16* vk = (_Float16*)alloc((size_t)CB * CK * CE * 2);
  _Float16* qu = (_Float16*)alloc((size_t)CB * CS * CE * 2);
  _Float16* qv = (_Float16*)alloc((size_t)CB * CS * CE * 2);
  _Float16* kf = (_Float16*)alloc((size_t)CB * CK * CE * 2);
  _Float16* vT = (_Float16*)alloc((size_t)CB * CE * CK * 2);
  _Float16* rf = (_Float16*)alloc((size_t)CK * CE * 2);
  _Float16* o_h = (_Float16*)alloc((size_t)CB * CS * CE * 2);
  float*    hbuf = (float*)alloc((size_t)CB * CS * CE * 4);
  _Float16* hn = vk;    // alias: vk dead after Q/K/V projections
  _Float16* ff = o_h;   // alias: o dead after Wo GEMM

  const dim3 blk(256);
  const int M1 = CB * CS;  // 4096
  const int M2 = CB * CK;  // 8192

  // one-time conversions (idempotent, graph-capture safe)
  pe_kernel<<<dim3((CK * CE) / 256), blk, 0, stream>>>(pe);
  conv_kernel<<<dim3((M1 * COBS) / 256), blk, 0, stream>>>(obs, obs_h);
  convT_kernel<<<dim3((COBS * CE) / 256), blk, 0, stream>>>(W_enc, WencT, COBS,
                                                            CE);
  for (int i = 0; i < CL; ++i) {
    const size_t wo = i * WPL;
    const dim3 g((CE * CE) / 256);
    convT_kernel<<<g, blk, 0, stream>>>(Wq + wo, WqT + wo, CE, CE);
    convT_kernel<<<g, blk, 0, stream>>>(Wk + wo, WkT + wo, CE, CE);
    convT_kernel<<<g, blk, 0, stream>>>(Wv + wo, WvT + wo, CE, CE);
    convT_kernel<<<g, blk, 0, stream>>>(Wr + wo, WrT + wo, CE, CE);
    convT_kernel<<<g, blk, 0, stream>>>(Wo + wo, WoT + wo, CE, CE);
    convT_kernel<<<g, blk, 0, stream>>>(W1 + wo, W1T + wo, CE, CE);
    convT_kernel<<<g, blk, 0, stream>>>(W2 + wo, W2T + wo, CE, CE);
  }

  // encoder: x = obs @ W_enc + b_enc   (M=4096, K=128, N=512)
  gemm_wmma<5><<<dim3(CE / 64, M1 / 64), blk, 0, stream>>>(
      obs_h, WencT, b_enc, nullptr, nullptr, x, nullptr, M1, CE, COBS, M1, M1,
      0, 1);

  for (int i = 0; i < CL; ++i) {
    const size_t lo = (size_t)i * CE;  // per-layer vector offset
    const size_t wo = i * WPL;         // per-layer matrix offset

    ln_concat_kernel<<<dim3(M2), blk, 0, stream>>>(x, memories, ln1_s + lo,
                                                   ln1_b + lo, vk, i);

    // Q projection -> qu = q+bq+u, qv = q+bq+v (A = vk rows [b*1024+512+s])
    gemm_wmma<1><<<dim3(CE / 64, M1 / 64), blk, 0, stream>>>(
        vk, WqT + wo, bq + lo, ub + lo, vb + lo, qu, qv, M1, CE, CE, CS, CK,
        CMEM, 1);
    // K projection (f16, [B,K,E])
    gemm_wmma<0><<<dim3(CE / 64, M2 / 64), blk, 0, stream>>>(
        vk, WkT + wo, bk + lo, nullptr, nullptr, kf, nullptr, M2, CE, CE, M2,
        M2, 0, 1);
    // V projection, transposed f16 out [B, E, K]
    gemm_wmma<2><<<dim3(CE / 64, M2 / 64), blk, 0, stream>>>(
        vk, WvT + wo, bv + lo, nullptr, nullptr, vT, nullptr, M2, CE, CE, M2,
        M2, 0, CK);
    // r = pe @ Wr (no bias), [K, E] f16
    gemm_wmma<0><<<dim3(CE / 64, CK / 64), blk, 0, stream>>>(
        pe, WrT + wo, nullptr, nullptr, nullptr, rf, nullptr, CK, CE, CE, CK,
        CK, 0, 1);

    // fused flash attention -> o [B,S,E] f16
    attn_kernel<<<dim3((CB * CH * (CS / 16)) / 8), blk, 0, stream>>>(
        qu, qv, kf, vT, rf, o_h);

    // h = x + o @ Wo + bo   (f32)
    gemm_wmma<3><<<dim3(CE / 64, M1 / 64), blk, 0, stream>>>(
        o_h, WoT + wo, bo + lo, x, nullptr, hbuf, nullptr, M1, CE, CE, M1, M1,
        0, 1);

    ln_kernel<<<dim3(M1), blk, 0, stream>>>(hbuf, ln2_s + lo, ln2_b + lo, hn);

    // ff = gelu(hn @ W1 + b1)  (f16)
    gemm_wmma<4><<<dim3(CE / 64, M1 / 64), blk, 0, stream>>>(
        hn, W1T + wo, b1 + lo, nullptr, nullptr, ff, nullptr, M1, CE, CE, M1,
        M1, 0, 1);
    // x_new = h + ff @ W2 + b2  (last layer writes straight into d_out)
    float* xdst = (i == CL - 1) ? (float*)d_out : x;
    gemm_wmma<3><<<dim3(CE / 64, M1 / 64), blk, 0, stream>>>(
        ff, W2T + wo, b2 + lo, hbuf, nullptr, xdst, nullptr, M1, CE, CE, M1,
        M1, 0, 1);
  }
}
